// DisentangledSelfAttention_7954279432210
// MI455X (gfx1250) — compile-verified, matrix operand order checked by reading, not run on hardware
//
#include <hip/hip_runtime.h>
#include <math.h>

// ---------------- problem constants (B=1) ----------------
constexpr int kS  = 2048;   // sequence length
constexpr int kD  = 1024;   // hidden dim
constexpr int kH  = 16;     // heads
constexpr int kDh = 64;     // head dim
constexpr int kP  = 512;    // 2*SPAN
constexpr float kInvScale = 0.07216878364870323f; // 1/sqrt(64*3)

typedef __attribute__((ext_vector_type(16))) __bf16 v16bf;
typedef __attribute__((ext_vector_type(8)))  __bf16 v8bf;
typedef __attribute__((ext_vector_type(8)))  float  v8f;

union FragBF { v16bf v; v8bf h[2]; };

static __device__ __forceinline__ v8f wmma_bf16(v16bf a, v16bf b, v8f c) {
  // D = A(16x32 bf16) * B(32x16 bf16) + C(16x16 f32)
  return __builtin_amdgcn_wmma_f32_16x16x32_bf16(false, a, false, b, (short)0, c,
                                                 false, false);
}

// ---------------- fp32 -> bf16 convert ----------------
__global__ void k_cvt(const float* __restrict__ src, __bf16* __restrict__ dst, int n) {
  int i = blockIdx.x * blockDim.x + threadIdx.x;
  int stride = gridDim.x * blockDim.x;
  for (; i < n; i += stride) dst[i] = (__bf16)src[i];
}

// transpose-convert 1024x1024 weight: dst[n*kD + k] = (bf16)src[k*kD + n]
__global__ void k_cvtT(const float* __restrict__ src, __bf16* __restrict__ dst) {
  __shared__ float tile[32][33];
  const int bx = blockIdx.x * 32, by = blockIdx.y * 32;
  const int tx = threadIdx.x, ty = threadIdx.y;           // blockDim (32,8)
  for (int r = ty; r < 32; r += 8) tile[r][tx] = src[(size_t)(by + r) * kD + bx + tx];
  __syncthreads();
  for (int r = ty; r < 32; r += 8)
    dst[(size_t)(bx + r) * kD + by + tx] = (__bf16)tile[tx][r];
}

// ---------------- C = A(Mx1024) * Wt^T + bias ----------------
// One wave computes a 16x64 strip: A fragment reused across 4 B tiles.
// All 5 fragments are loaded up-front (distinct live values) so the scheduler
// can clause the loads and drain with partial s_wait_loadcnt instead of
// serializing load->wmma.
// Output head-major: dstH[(h*M + m)*64 + d], h = n/64, d = n%64 (split_heads).
__global__ void k_gemm_bias(const __bf16* __restrict__ A, const __bf16* __restrict__ Wt,
                            const float* __restrict__ bias, __bf16* __restrict__ dstH,
                            int M) {
  const int lane  = threadIdx.x;
  const int tn    = blockIdx.x * 64;
  const int tm    = blockIdx.y * 16;
  const int l15   = lane & 15;
  const int khalf = (lane >> 4) * 8;       // K split between lane halves
  const int hi8   = (lane >> 4) << 3;      // C-layout row offset
  const __bf16* arow = A  + (size_t)(tm + l15) * kD;
  const __bf16* bcol = Wt + (size_t)(tn + l15) * kD;
  v8f acc[4];
  v8f zero = {};
#pragma unroll
  for (int j = 0; j < 4; ++j) acc[j] = zero;
  for (int kk = 0; kk < kD; kk += 32) {
    FragBF a, b0, b1, b2, b3;
    a.h[0]  = *(const v8bf*)(arow + kk + khalf);
    a.h[1]  = *(const v8bf*)(arow + kk + 16 + khalf);
    b0.h[0] = *(const v8bf*)(bcol + 0 * 16 * kD + kk + khalf);
    b0.h[1] = *(const v8bf*)(bcol + 0 * 16 * kD + kk + 16 + khalf);
    b1.h[0] = *(const v8bf*)(bcol + 1 * 16 * kD + kk + khalf);
    b1.h[1] = *(const v8bf*)(bcol + 1 * 16 * kD + kk + 16 + khalf);
    b2.h[0] = *(const v8bf*)(bcol + 2 * 16 * kD + kk + khalf);
    b2.h[1] = *(const v8bf*)(bcol + 2 * 16 * kD + kk + 16 + khalf);
    b3.h[0] = *(const v8bf*)(bcol + 3 * 16 * kD + kk + khalf);
    b3.h[1] = *(const v8bf*)(bcol + 3 * 16 * kD + kk + 16 + khalf);
    acc[0] = wmma_bf16(a.v, b0.v, acc[0]);
    acc[1] = wmma_bf16(a.v, b1.v, acc[1]);
    acc[2] = wmma_bf16(a.v, b2.v, acc[2]);
    acc[3] = wmma_bf16(a.v, b3.v, acc[3]);
  }
  const int m0 = tm + hi8;
#pragma unroll
  for (int j = 0; j < 4; ++j) {
    const int n  = tn + j * 16 + l15;
    const float bv = bias[n];
    const int hh = n >> 6, dd = n & 63;
#pragma unroll
    for (int r = 0; r < 8; ++r)
      dstH[((size_t)hh * M + m0 + r) * kDh + dd] = (__bf16)(acc[j][r] + bv);
  }
}

// Same GEMM but output transposed per head: dstT[(h*64 + d)*M + m] (for V).
// Epilogue rows are consecutive in m -> contiguous v8bf stores.
__global__ void k_gemm_biasT(const __bf16* __restrict__ A, const __bf16* __restrict__ Wt,
                             const float* __restrict__ bias, __bf16* __restrict__ dstT,
                             int M) {
  const int lane  = threadIdx.x;
  const int tn    = blockIdx.x * 64;
  const int tm    = blockIdx.y * 16;
  const int l15   = lane & 15;
  const int khalf = (lane >> 4) * 8;
  const int hi8   = (lane >> 4) << 3;
  const __bf16* arow = A  + (size_t)(tm + l15) * kD;
  const __bf16* bcol = Wt + (size_t)(tn + l15) * kD;
  v8f acc[4];
  v8f zero = {};
#pragma unroll
  for (int j = 0; j < 4; ++j) acc[j] = zero;
  for (int kk = 0; kk < kD; kk += 32) {
    FragBF a, b0, b1, b2, b3;
    a.h[0]  = *(const v8bf*)(arow + kk + khalf);
    a.h[1]  = *(const v8bf*)(arow + kk + 16 + khalf);
    b0.h[0] = *(const v8bf*)(bcol + 0 * 16 * kD + kk + khalf);
    b0.h[1] = *(const v8bf*)(bcol + 0 * 16 * kD + kk + 16 + khalf);
    b1.h[0] = *(const v8bf*)(bcol + 1 * 16 * kD + kk + khalf);
    b1.h[1] = *(const v8bf*)(bcol + 1 * 16 * kD + kk + 16 + khalf);
    b2.h[0] = *(const v8bf*)(bcol + 2 * 16 * kD + kk + khalf);
    b2.h[1] = *(const v8bf*)(bcol + 2 * 16 * kD + kk + 16 + khalf);
    b3.h[0] = *(const v8bf*)(bcol + 3 * 16 * kD + kk + khalf);
    b3.h[1] = *(const v8bf*)(bcol + 3 * 16 * kD + kk + 16 + khalf);
    acc[0] = wmma_bf16(a.v, b0.v, acc[0]);
    acc[1] = wmma_bf16(a.v, b1.v, acc[1]);
    acc[2] = wmma_bf16(a.v, b2.v, acc[2]);
    acc[3] = wmma_bf16(a.v, b3.v, acc[3]);
  }
  const int m0 = tm + hi8;
#pragma unroll
  for (int j = 0; j < 4; ++j) {
    const int n  = tn + j * 16 + l15;
    const float bv = bias[n];
    const int hh = n >> 6, dd = n & 63;
    v8bf o;
#pragma unroll
    for (int r = 0; r < 8; ++r) o[r] = (__bf16)(acc[j][r] + bv);
    *(v8bf*)(dstT + ((size_t)hh * kDh + dd) * M + m0) = o;
  }
}

// ---------------- rel = Ah(H,M,64) * Bh(H,512,64)^T, pre-scaled, bf16 ----------------
// One wave computes a 16x64 strip (A fragments loaded once, K = 64 total).
__global__ void k_relscore(const __bf16* __restrict__ Ah, const __bf16* __restrict__ Bh,
                           __bf16* __restrict__ dst, int M) {
  const int lane  = threadIdx.x;
  const int tn    = blockIdx.x * 64;
  const int tm    = blockIdx.y * 16;
  const int hh    = blockIdx.z;
  const int l15   = lane & 15;
  const int khalf = (lane >> 4) * 8;
  const int hi8   = (lane >> 4) << 3;
  const __bf16* arow = Ah + ((size_t)hh * M  + tm + l15) * kDh;
  const __bf16* brow = Bh + ((size_t)hh * kP + tn + l15) * kDh;
  FragBF a0, a1;
  a0.h[0] = *(const v8bf*)(arow + khalf);
  a0.h[1] = *(const v8bf*)(arow + 16 + khalf);
  a1.h[0] = *(const v8bf*)(arow + 32 + khalf);
  a1.h[1] = *(const v8bf*)(arow + 48 + khalf);
  const int m0 = tm + hi8;
#pragma unroll
  for (int j = 0; j < 4; ++j) {
    const __bf16* bj = brow + (size_t)j * 16 * kDh;
    FragBF b0, b1;
    b0.h[0] = *(const v8bf*)(bj + khalf);
    b0.h[1] = *(const v8bf*)(bj + 16 + khalf);
    b1.h[0] = *(const v8bf*)(bj + 32 + khalf);
    b1.h[1] = *(const v8bf*)(bj + 48 + khalf);
    v8f acc = {};
    acc = wmma_bf16(a0.v, b0.v, acc);
    acc = wmma_bf16(a1.v, b1.v, acc);
#pragma unroll
    for (int r = 0; r < 8; ++r)
      dst[((size_t)hh * M + m0 + r) * kP + tn + j * 16 + l15] =
          (__bf16)(acc[r] * kInvScale);
  }
}

// ---------------- fused flash attention with gathered disentangled bias ----------------
// One wave32 per (head, 16-row query tile). Key tiles of 32.
// V is pre-transposed (H,64,S) so PV B-fragments are contiguous global loads.
__global__ void k_attn(const __bf16* __restrict__ Qh, const __bf16* __restrict__ Kh,
                       const __bf16* __restrict__ Vt, const __bf16* __restrict__ c2p,
                       const __bf16* __restrict__ p2c, const int* __restrict__ ids,
                       const int* __restrict__ ids_t, float* __restrict__ out) {
  const int lane  = threadIdx.x;
  const int hh    = blockIdx.y;
  const int qb    = blockIdx.x * 16;
  const int l15   = lane & 15;
  const int hi8   = (lane >> 4) << 3;      // row offset for C-fragment rows
  const int khalf = (lane >> 4) * 8;

  __shared__ __align__(16) __bf16 Plds[16 * 32];

  // Q fragments: A layout, row = l15, K = 0..31 and 32..63
  FragBF qa0, qa1;
  {
    const __bf16* qrow = Qh + ((size_t)hh * kS + qb + l15) * kDh;
    qa0.h[0] = *(const v8bf*)(qrow + khalf);
    qa0.h[1] = *(const v8bf*)(qrow + 16 + khalf);
    qa1.h[0] = *(const v8bf*)(qrow + 32 + khalf);
    qa1.h[1] = *(const v8bf*)(qrow + 48 + khalf);
  }

  float rmax[8], rsum[8];
  v8f acc[4];
  v8f zero = {};
#pragma unroll
  for (int r = 0; r < 8; ++r) { rmax[r] = -3.0e38f; rsum[r] = 0.f; }
#pragma unroll
  for (int d = 0; d < 4; ++d) acc[d] = zero;

  const size_t idbase = (size_t)hh * kS * kS;

  for (int t = 0; t < kS; t += 32) {
    // K fragments: B layout, col = key (l15), K = feature (contiguous)
    const __bf16* krow0 = Kh + ((size_t)hh * kS + t + l15) * kDh;
    const __bf16* krow1 = krow0 + 16 * kDh;
    if (t + 32 < kS) __builtin_prefetch(krow0 + 32 * kDh, 0, 0);  // global_prefetch_b8
    FragBF b0a, b0b, b1a, b1b;
    b0a.h[0] = *(const v8bf*)(krow0 + khalf);
    b0a.h[1] = *(const v8bf*)(krow0 + 16 + khalf);
    b0b.h[0] = *(const v8bf*)(krow0 + 32 + khalf);
    b0b.h[1] = *(const v8bf*)(krow0 + 48 + khalf);
    b1a.h[0] = *(const v8bf*)(krow1 + khalf);
    b1a.h[1] = *(const v8bf*)(krow1 + 16 + khalf);
    b1b.h[0] = *(const v8bf*)(krow1 + 32 + khalf);
    b1b.h[1] = *(const v8bf*)(krow1 + 48 + khalf);

    v8f s0 = zero, s1 = zero;
    s0 = wmma_bf16(qa0.v, b0a.v, s0);
    s0 = wmma_bf16(qa1.v, b0b.v, s0);
    s1 = wmma_bf16(qa0.v, b1a.v, s1);
    s1 = wmma_bf16(qa1.v, b1b.v, s1);

    // bias gathers (L2-resident c2p/p2c) + online softmax, per C-fragment row
    const int t0 = t + l15, t1 = t + 16 + l15;
#pragma unroll
    for (int r = 0; r < 8; ++r) {
      const int srow = qb + hi8 + r;
      const size_t rowoff = idbase + (size_t)srow * kS;
      float v0 = s0[r] * kInvScale + (float)c2p[ids[rowoff + t0]]
                                   + (float)p2c[ids_t[idbase + (size_t)t0 * kS + srow]];
      float v1 = s1[r] * kInvScale + (float)c2p[ids[rowoff + t1]]
                                   + (float)p2c[ids_t[idbase + (size_t)t1 * kS + srow]];
      float tmax = fmaxf(v0, v1);
#pragma unroll
      for (int off = 1; off < 16; off <<= 1)
        tmax = fmaxf(tmax, __shfl_xor(tmax, off, 32));
      const float mnew = fmaxf(rmax[r], tmax);
      const float fscale = __expf(rmax[r] - mnew);
      const float p0 = __expf(v0 - mnew);
      const float p1 = __expf(v1 - mnew);
      float ps = p0 + p1;
#pragma unroll
      for (int off = 1; off < 16; off <<= 1) ps += __shfl_xor(ps, off, 32);
      rsum[r] = rsum[r] * fscale + ps;
      rmax[r] = mnew;
#pragma unroll
      for (int d = 0; d < 4; ++d) acc[d][r] *= fscale;
      Plds[(hi8 + r) * 32 + l15]      = (__bf16)p0;
      Plds[(hi8 + r) * 32 + 16 + l15] = (__bf16)p1;
    }
    __syncthreads();

    // P as A-fragment (16x32), re-layout through LDS
    FragBF pa;
    pa.h[0] = *(const v8bf*)(Plds + l15 * 32 + khalf);
    pa.h[1] = *(const v8bf*)(Plds + l15 * 32 + 16 + khalf);

    // V as B-fragments: contiguous 16B loads from transposed Vt (H,64,S)
#pragma unroll
    for (int dt = 0; dt < 4; ++dt) {
      const __bf16* vrow = Vt + ((size_t)hh * kDh + dt * 16 + l15) * kS + t;
      FragBF vb;
      vb.h[0] = *(const v8bf*)(vrow + khalf);
      vb.h[1] = *(const v8bf*)(vrow + 16 + khalf);
      acc[dt] = wmma_bf16(pa.v, vb.v, acc[dt]);
    }
    __syncthreads();
  }

  // epilogue: normalize and scatter to (S, D) fp32, undoing split_heads
#pragma unroll
  for (int r = 0; r < 8; ++r) {
    const int srow = qb + hi8 + r;
    const float inv = 1.0f / rsum[r];
#pragma unroll
    for (int dt = 0; dt < 4; ++dt)
      out[(size_t)srow * kD + hh * kDh + dt * 16 + l15] = acc[dt][r] * inv;
  }
}

// ---------------- host side ----------------
extern "C" void kernel_launch(void* const* d_in, const int* in_sizes, int n_in,
                              void* d_out, int out_size, void* d_ws, size_t ws_size,
                              hipStream_t stream) {
  (void)in_sizes; (void)n_in; (void)out_size; (void)ws_size;
  const float* hs  = (const float*)d_in[0];
  const float* rel = (const float*)d_in[1];
  const float* Wq  = (const float*)d_in[2];
  const float* bq  = (const float*)d_in[3];
  const float* Wk  = (const float*)d_in[4];
  const float* bk  = (const float*)d_in[5];
  const float* Wv  = (const float*)d_in[6];
  const float* bv  = (const float*)d_in[7];
  const int*   ids   = (const int*)d_in[8];
  const int*   ids_t = (const int*)d_in[9];
  float* out = (float*)d_out;

  char* ws = (char*)d_ws;
  auto carve = [&](size_t bytes) {
    char* p = ws; ws += (bytes + 255) & ~(size_t)255; return p;
  };
  __bf16* Xbf  = (__bf16*)carve((size_t)kS * kD * 2);
  __bf16* Pbf  = (__bf16*)carve((size_t)kP * kD * 2);
  __bf16* Wqt  = (__bf16*)carve((size_t)kD * kD * 2);
  __bf16* Wkt  = (__bf16*)carve((size_t)kD * kD * 2);
  __bf16* Wvt  = (__bf16*)carve((size_t)kD * kD * 2);
  __bf16* Qh   = (__bf16*)carve((size_t)kH * kS * kDh * 2);
  __bf16* Kh   = (__bf16*)carve((size_t)kH * kS * kDh * 2);
  __bf16* Vt   = (__bf16*)carve((size_t)kH * kDh * kS * 2);  // transposed V
  __bf16* pQh  = (__bf16*)carve((size_t)kH * kP * kDh * 2);
  __bf16* pKh  = (__bf16*)carve((size_t)kH * kP * kDh * 2);
  __bf16* c2p  = (__bf16*)carve((size_t)kH * kS * kP * 2);   // 32 MB, L2-resident
  __bf16* p2c  = (__bf16*)carve((size_t)kH * kS * kP * 2);   // 32 MB, L2-resident

  // 1) convert inputs
  k_cvt<<<1024, 256, 0, stream>>>(hs, Xbf, kS * kD);
  k_cvt<<<256, 256, 0, stream>>>(rel, Pbf, kP * kD);
  dim3 tg(kD / 32, kD / 32), tb(32, 8);
  k_cvtT<<<tg, tb, 0, stream>>>(Wq, Wqt);
  k_cvtT<<<tg, tb, 0, stream>>>(Wk, Wkt);
  k_cvtT<<<tg, tb, 0, stream>>>(Wv, Wvt);

  // 2) projections (WMMA, 16x64 strip per wave)
  dim3 gX(kD / 64, kS / 16);
  k_gemm_bias <<<gX, 32, 0, stream>>>(Xbf, Wqt, bq, Qh, kS);
  k_gemm_bias <<<gX, 32, 0, stream>>>(Xbf, Wkt, bk, Kh, kS);
  k_gemm_biasT<<<gX, 32, 0, stream>>>(Xbf, Wvt, bv, Vt, kS);
  dim3 gPp(kD / 64, kP / 16);
  k_gemm_bias<<<gPp, 32, 0, stream>>>(Pbf, Wqt, bq, pQh, kP);
  k_gemm_bias<<<gPp, 32, 0, stream>>>(Pbf, Wkt, bk, pKh, kP);

  // 3) relative score tables (WMMA, pre-scaled by 1/sqrt(192))
  dim3 gR(kP / 64, kS / 16, kH);
  k_relscore<<<gR, 32, 0, stream>>>(Qh, pKh, c2p, kS);   // c2p = Q . pos_k^T
  k_relscore<<<gR, 32, 0, stream>>>(Kh, pQh, p2c, kS);   // p2c = K . pos_q^T

  // 4) fused attention
  dim3 gA(kS / 16, kH);
  k_attn<<<gA, 32, 0, stream>>>(Qh, Kh, Vt, c2p, p2c, ids, ids_t, out);
}
